// FactorizedAttention_21440476742181
// MI455X (gfx1250) — matrix operand order repaired and checked by reading, not compiled, not run on hardware
//
#include <hip/hip_runtime.h>

#define BATCH 32
#define NTOK  785
#define NPAD  800       // token dim padded to multiple of 32 for WMMA K-loop
#define DIMX  768
#define HEADS 8
#define HDIM  96
#define GH    28
#define GW    28
#define QSCALE 0.10206207261596577f   // 96^-0.5

typedef __attribute__((ext_vector_type(16))) __bf16 v16bf;
typedef __attribute__((ext_vector_type(8)))  float  v8f;
typedef unsigned short u16;
typedef unsigned int   u32;

union Frag { v16bf v; u32 u[8]; };

__device__ __forceinline__ u16 f32_to_bf16(float x) {
  u32 u = __float_as_uint(x);
  return (u16)((u + 0x7FFFu + ((u >> 16) & 1u)) >> 16);
}

// A-matrix 16x32 bf16 fragment (ISA 7.12.2): lane holds row M=lane&15;
// pair p covers K = ((p>>2)<<4) + 8*(lane>>4) + 2*(p&3)
// => per-lane bytes [0..15] and [32..47] of the row (compiler emits 2x b128)
__device__ __forceinline__ void load_a(Frag& f, const u16* p, int khiA) {
#pragma unroll
  for (int q = 0; q < 8; ++q) {
    int k = ((q >> 2) << 4) + khiA + ((q & 3) << 1);
    f.u[q] = *(const u32*)(p + k);
  }
}
__device__ __forceinline__ void zero_frag(Frag& f) {
#pragma unroll
  for (int q = 0; q < 8; ++q) f.u[q] = 0u;
}
// B-matrix 32x16 bf16 fragment: lane holds col N=lane&15; VGPR p covers
// K = ((p>>2)<<4) + 8*(lane>>4) + 2*(p&3) .. +1   (fed from B^T stored row-major, K contiguous)
__device__ __forceinline__ void load_b(Frag& f, const u16* p, int khiB) {
#pragma unroll
  for (int q = 0; q < 8; ++q) f.u[q] = *(const u32*)(p + ((q >> 2) << 4) + (khiB >> 1) + ((q & 3) << 1));
}
__device__ __forceinline__ v8f wmma_step(const Frag& a, const Frag& b, v8f c) {
  return __builtin_amdgcn_wmma_f32_16x16x32_bf16(false, a.v, false, b.v,
                                                 (short)0, c, false, false);
}

// ---------------- conversion kernels ----------------
__global__ __launch_bounds__(256) void cvt_f32_bf16(const float* __restrict__ in,
                                                    u16* __restrict__ out, int n) {
  int i = blockIdx.x * 256 + threadIdx.x;
  if (i < n) out[i] = f32_to_bf16(in[i]);
}
// in: rows x cols fp32 row-major -> out: cols x rows bf16 (transposed)
__global__ __launch_bounds__(256) void transpose_bf16(const float* __restrict__ in,
                                                      u16* __restrict__ out,
                                                      int rows, int cols) {
  int i = blockIdx.x * 256 + threadIdx.x;
  if (i >= rows * cols) return;
  int r = i / cols, c = i % cols;
  out[(size_t)c * rows + r] = f32_to_bf16(in[(size_t)r * cols + c]);
}

// ---------------- QKV GEMM:  (25120,768) x (768,2304), 32x64 block per wave ----------------
// register-blocked 2 M-tiles x 4 N-tiles: 6 KB operands per 8 WMMAs (~21 flop/B)
__global__ __launch_bounds__(128) void qkv_gemm(
    const u16* __restrict__ xb, const u16* __restrict__ wqt,
    const float* __restrict__ bias,
    float* __restrict__ qf, u16* __restrict__ qb,
    float* __restrict__ kT, float* __restrict__ vf, u16* __restrict__ vT) {
  const int lane = threadIdx.x & 31;
  const int wave = threadIdx.x >> 5;
  const int NG = (3 * DIMX) / 64;                 // 36 N-groups of 4 tiles
  const int grp = blockIdx.x * 4 + wave;          // 785*36 groups, grid exact
  const int gm = grp / NG, gn = grp % NG;
  const int m0 = gm * 32, n0 = gn * 64;
  const int lr = lane & 15;
  const int khiA = (lane >> 4) << 3, khiB = (lane >> 4) << 4;
  const u16* ap0 = xb + (size_t)(m0 + lr) * DIMX;
  const u16* ap1 = xb + (size_t)(m0 + 16 + lr) * DIMX;
  const u16* bp[4];
#pragma unroll
  for (int j = 0; j < 4; ++j) bp[j] = wqt + (size_t)(n0 + j * 16 + lr) * DIMX;
  v8f z = {};
  v8f acc[2][4];
#pragma unroll
  for (int i = 0; i < 2; ++i)
#pragma unroll
    for (int j = 0; j < 4; ++j) acc[i][j] = z;
  for (int kk = 0; kk < DIMX; kk += 32) {
    Frag a0, a1, b0, b1, b2, b3;
    load_a(a0, ap0 + kk, khiA);
    load_a(a1, ap1 + kk, khiA);
    load_b(b0, bp[0] + kk, khiB);
    load_b(b1, bp[1] + kk, khiB);
    load_b(b2, bp[2] + kk, khiB);
    load_b(b3, bp[3] + kk, khiB);
    acc[0][0] = wmma_step(a0, b0, acc[0][0]);
    acc[0][1] = wmma_step(a0, b1, acc[0][1]);
    acc[0][2] = wmma_step(a0, b2, acc[0][2]);
    acc[0][3] = wmma_step(a0, b3, acc[0][3]);
    acc[1][0] = wmma_step(a1, b0, acc[1][0]);
    acc[1][1] = wmma_step(a1, b1, acc[1][1]);
    acc[1][2] = wmma_step(a1, b2, acc[1][2]);
    acc[1][3] = wmma_step(a1, b3, acc[1][3]);
  }
  // epilogue: bias + scatter to q / kT / v layouts
#pragma unroll
  for (int nj = 0; nj < 4; ++nj) {
    const int col = n0 + nj * 16 + lr;
    const float bv = bias[col];
    const int s = col / DIMX;                     // 0=q 1=k 2=v
    const int r = col % DIMX;
    const int hh = r / HDIM, dd = r % HDIM;
#pragma unroll
    for (int mi = 0; mi < 2; ++mi) {
#pragma unroll
      for (int i = 0; i < 8; ++i) {
        int row = m0 + mi * 16 + khiA + i;
        int bi = row / NTOK, tok = row % NTOK;
        float val = acc[mi][nj][i] + bv;
        size_t bh = (size_t)bi * HEADS + hh;
        if (s == 0) {
          size_t o = (bh * NTOK + tok) * HDIM + dd;
          qf[o] = val;
          qb[o] = f32_to_bf16(val * QSCALE);
        } else if (s == 1) {
          kT[(bh * HDIM + dd) * NTOK + tok] = val;
        } else {
          vf[(bh * NTOK + tok) * HDIM + dd] = val;
          vT[(bh * HDIM + dd) * NPAD + tok] = f32_to_bf16(val);
        }
      }
    }
  }
}

// zero the K-pad columns [785,800) of vT
__global__ __launch_bounds__(256) void pad_vt(u16* __restrict__ vT) {
  int i = blockIdx.x * 256 + threadIdx.x;
  const int PAD = NPAD - NTOK;                    // 15
  if (i >= BATCH * HEADS * HDIM * PAD) return;
  int p = i % PAD, ch = i / PAD;
  vT[(size_t)ch * NPAD + NTOK + p] = 0;
}

// ---------------- softmax over tokens per (b,h,d) row; write bf16 padded ----------------
__global__ __launch_bounds__(256) void softmax_k(const float* __restrict__ kT,
                                                 u16* __restrict__ ksm) {
  const int lane = threadIdx.x & 31;
  const int wave = threadIdx.x >> 5;
  const int row = blockIdx.x * 8 + wave;          // < 24576 (grid exact)
  const float* src = kT + (size_t)row * NTOK;
  u16* dst = ksm + (size_t)row * NPAD;
  float vals[25];
  int cnt = 0;
  float m = -3.402823466e38f;
  for (int t = lane; t < NTOK; t += 32) { vals[cnt] = src[t]; m = fmaxf(m, vals[cnt]); ++cnt; }
#pragma unroll
  for (int o = 16; o > 0; o >>= 1) m = fmaxf(m, __shfl_xor(m, o, 32));
  float s = 0.f;
  for (int i = 0; i < cnt; ++i) { vals[i] = __expf(vals[i] - m); s += vals[i]; }
#pragma unroll
  for (int o = 16; o > 0; o >>= 1) s += __shfl_xor(s, o, 32);
  float inv = 1.f / s;
  cnt = 0;
  for (int t = lane; t < NPAD; t += 32)
    dst[t] = (t < NTOK) ? f32_to_bf16(vals[cnt++] * inv) : (u16)0;
}

// ---------------- attn = k_sm^T(96x800) @ v(800x96) per (b,h); write attn^T bf16 ----------------
__global__ __launch_bounds__(256) void attn_gemm(const u16* __restrict__ ksm,
                                                 const u16* __restrict__ vT,
                                                 u16* __restrict__ attnT) {
  const int bh = blockIdx.x;
  const int lane = threadIdx.x & 31;
  const int wave = threadIdx.x >> 5;
  const int lr = lane & 15;
  const int khiA = (lane >> 4) << 3, khiB = (lane >> 4) << 4;
  const u16* abase = ksm + (size_t)bh * HDIM * NPAD;
  const u16* bbase = vT  + (size_t)bh * HDIM * NPAD;
  u16* obase = attnT + (size_t)bh * HDIM * HDIM;
  for (int t = wave; t < 36; t += 8) {            // 6x6 tiles, wave-uniform loop
    int tm = t / 6, tn = t % 6;
    const u16* aptr = abase + (size_t)(tm * 16 + lr) * NPAD;
    const u16* bptr = bbase + (size_t)(tn * 16 + lr) * NPAD;
    v8f acc = {};
    for (int kk = 0; kk < NPAD; kk += 32) {
      Frag a, b;
      load_a(a, aptr + kk, khiA);
      load_b(b, bptr + kk, khiB);
      acc = wmma_step(a, b, acc);
    }
    int e = tn * 16 + lr;
#pragma unroll
    for (int i = 0; i < 8; ++i) {
      int d = tm * 16 + khiA + i;
      obase[(size_t)e * HDIM + d] = f32_to_bf16(acc[i]);   // store transposed
    }
  }
}

// ---------------- CRPE depthwise conv on v image ----------------
__global__ __launch_bounds__(256) void crpe_conv(
    const float* __restrict__ vf,
    const float* __restrict__ w3, const float* __restrict__ b3,
    const float* __restrict__ w5, const float* __restrict__ b5,
    const float* __restrict__ w7, const float* __restrict__ b7,
    float* __restrict__ rp) {
  int idx = blockIdx.x * 256 + threadIdx.x;
  const int total = BATCH * HEADS * HDIM * GH * GW;
  if (idx >= total) return;
  int xx = idx % GW;
  int yy = (idx / GW) % GH;
  int dd = (idx / (GW * GH)) % HDIM;
  int hh = (idx / (GW * GH * HDIM)) % HEADS;
  int bi = idx / (GW * GH * HDIM * HEADS);
  int k; const float* w; const float* b; int c;
  if (hh < 2)      { k = 3; w = w3; b = b3; c = hh * HDIM + dd; }
  else if (hh < 5) { k = 5; w = w5; b = b5; c = (hh - 2) * HDIM + dd; }
  else             { k = 7; w = w7; b = b7; c = (hh - 5) * HDIM + dd; }
  const float* wk = w + (size_t)c * k * k;
  int rad = k >> 1;
  float acc = b[c];
  const float* vb = vf + (((size_t)bi * HEADS + hh) * NTOK + 1) * HDIM + dd;
  for (int ky = 0; ky < k; ++ky) {
    int iy = yy + ky - rad;
    if (iy < 0 || iy >= GH) continue;
    for (int kx = 0; kx < k; ++kx) {
      int ix = xx + kx - rad;
      if (ix < 0 || ix >= GW) continue;
      acc += wk[ky * k + kx] * vb[(size_t)(iy * GW + ix) * HDIM];
    }
  }
  rp[(((size_t)bi * HEADS + hh) * NTOK + 1 + yy * GW + xx) * HDIM + dd] = acc;
}

// ---------------- out1 = q_scaled @ attn + q*rp  per (b,h); write bf16 (b,n,768) ----------------
__global__ __launch_bounds__(256) void out1_gemm(
    const u16* __restrict__ qb, const u16* __restrict__ attnT,
    const float* __restrict__ qf, const float* __restrict__ rp,
    u16* __restrict__ out1) {
  const int bh = blockIdx.y;
  const int bi = bh / HEADS, hh = bh % HEADS;
  const int lane = threadIdx.x & 31;
  const int wave = threadIdx.x >> 5;
  const int t = blockIdx.x * 8 + wave;
  if (t >= 50 * 6) return;                        // wave-uniform exit
  const int tm = t / 6, tn = t % 6;
  const int lr = lane & 15;
  const int khiA = (lane >> 4) << 3, khiB = (lane >> 4) << 4;
  const u16* abase = qb + (size_t)bh * NTOK * HDIM;
  const u16* bptr  = attnT + (size_t)bh * HDIM * HDIM + (size_t)(tn * 16 + lr) * HDIM;
  const int arow = tm * 16 + lr;
  const bool avalid = arow < NTOK;
  const u16* aptr = abase + (size_t)(avalid ? arow : 0) * HDIM;
  v8f acc = {};
  for (int kk = 0; kk < HDIM; kk += 32) {
    Frag a, b;
    if (avalid) load_a(a, aptr + kk, khiA); else zero_frag(a);
    load_b(b, bptr + kk, khiB);
    acc = wmma_step(a, b, acc);
  }
  const int e = tn * 16 + lr;
#pragma unroll
  for (int i = 0; i < 8; ++i) {
    int tok = tm * 16 + khiA + i;
    if (tok < NTOK) {
      size_t qo = ((size_t)bh * NTOK + tok) * HDIM + e;
      float add = (tok > 0) ? qf[qo] * rp[qo] : 0.f;   // rp row 0 (CLS) is zero
      out1[((size_t)bi * NTOK + tok) * DIMX + hh * HDIM + e] =
          f32_to_bf16(acc[i] + add);
    }
  }
}

// ---------------- final projection: (25120,768) x (768,768) + bias -> fp32 out ----------------
// register-blocked 2 M-tiles x 4 N-tiles per wave
__global__ __launch_bounds__(128) void proj_gemm(const u16* __restrict__ o1,
                                                 const u16* __restrict__ wot,
                                                 const float* __restrict__ bias,
                                                 float* __restrict__ y) {
  const int lane = threadIdx.x & 31;
  const int wave = threadIdx.x >> 5;
  const int NG = DIMX / 64;                       // 12 N-groups
  const int grp = blockIdx.x * 4 + wave;          // 785*12 groups, grid exact
  const int gm = grp / NG, gn = grp % NG;
  const int m0 = gm * 32, n0 = gn * 64;
  const int lr = lane & 15;
  const int khiA = (lane >> 4) << 3, khiB = (lane >> 4) << 4;
  const u16* ap0 = o1 + (size_t)(m0 + lr) * DIMX;
  const u16* ap1 = o1 + (size_t)(m0 + 16 + lr) * DIMX;
  const u16* bp[4];
#pragma unroll
  for (int j = 0; j < 4; ++j) bp[j] = wot + (size_t)(n0 + j * 16 + lr) * DIMX;
  v8f z = {};
  v8f acc[2][4];
#pragma unroll
  for (int i = 0; i < 2; ++i)
#pragma unroll
    for (int j = 0; j < 4; ++j) acc[i][j] = z;
  for (int kk = 0; kk < DIMX; kk += 32) {
    Frag a0, a1, b0, b1, b2, b3;
    load_a(a0, ap0 + kk, khiA);
    load_a(a1, ap1 + kk, khiA);
    load_b(b0, bp[0] + kk, khiB);
    load_b(b1, bp[1] + kk, khiB);
    load_b(b2, bp[2] + kk, khiB);
    load_b(b3, bp[3] + kk, khiB);
    acc[0][0] = wmma_step(a0, b0, acc[0][0]);
    acc[0][1] = wmma_step(a0, b1, acc[0][1]);
    acc[0][2] = wmma_step(a0, b2, acc[0][2]);
    acc[0][3] = wmma_step(a0, b3, acc[0][3]);
    acc[1][0] = wmma_step(a1, b0, acc[1][0]);
    acc[1][1] = wmma_step(a1, b1, acc[1][1]);
    acc[1][2] = wmma_step(a1, b2, acc[1][2]);
    acc[1][3] = wmma_step(a1, b3, acc[1][3]);
  }
#pragma unroll
  for (int nj = 0; nj < 4; ++nj) {
    const int col = n0 + nj * 16 + lr;
    const float bv = bias[col];
#pragma unroll
    for (int mi = 0; mi < 2; ++mi) {
#pragma unroll
      for (int i = 0; i < 8; ++i) {
        int row = m0 + mi * 16 + khiA + i;
        y[(size_t)row * DIMX + col] = acc[mi][nj][i] + bv;
      }
    }
  }
}

// ---------------- host launcher ----------------
extern "C" void kernel_launch(void* const* d_in, const int* in_sizes, int n_in,
                              void* d_out, int out_size, void* d_ws, size_t ws_size,
                              hipStream_t stream) {
  const float* x     = (const float*)d_in[0];
  const float* W_qkv = (const float*)d_in[1];
  const float* b_qkv = (const float*)d_in[2];
  const float* W_out = (const float*)d_in[3];
  const float* b_out = (const float*)d_in[4];
  const float* cw3   = (const float*)d_in[5];
  const float* cb3   = (const float*)d_in[6];
  const float* cw5   = (const float*)d_in[7];
  const float* cb5   = (const float*)d_in[8];
  const float* cw7   = (const float*)d_in[9];
  const float* cb7   = (const float*)d_in[10];
  float* y = (float*)d_out;

  char* ws = (char*)d_ws;
  size_t off = 0;
  auto alloc = [&](size_t bytes) {
    size_t o = off;
    off = (off + bytes + 255) & ~(size_t)255;
    return o;
  };
  const size_t NROWS = (size_t)BATCH * NTOK;      // 25120
  u16*   xb   = (u16*)  (ws + alloc(NROWS * DIMX * 2));
  u16*   wqt  = (u16*)  (ws + alloc((size_t)3 * DIMX * DIMX * 2));
  u16*   wot  = (u16*)  (ws + alloc((size_t)DIMX * DIMX * 2));
  float* qf   = (float*)(ws + alloc(NROWS * DIMX * 4));
  u16*   qb   = (u16*)  (ws + alloc(NROWS * DIMX * 2));
  float* kT   = (float*)(ws + alloc(NROWS * DIMX * 4));
  u16*   ksm  = (u16*)  (ws + alloc((size_t)BATCH * HEADS * HDIM * NPAD * 2));
  float* vf   = (float*)(ws + alloc(NROWS * DIMX * 4));
  u16*   vT   = (u16*)  (ws + alloc((size_t)BATCH * HEADS * HDIM * NPAD * 2));
  u16*   attnT= (u16*)  (ws + alloc((size_t)BATCH * HEADS * HDIM * HDIM * 2));
  float* rp   = (float*)(ws + alloc(NROWS * DIMX * 4));
  u16*   o1   = (u16*)  (ws + alloc(NROWS * DIMX * 2));
  (void)ws_size; (void)n_in; (void)in_sizes; (void)out_size;

  // 1) conversions
  {
    int n = (int)(NROWS * DIMX);
    cvt_f32_bf16<<<(n + 255) / 256, 256, 0, stream>>>(x, xb, n);
  }
  transpose_bf16<<<(3 * DIMX * DIMX + 255) / 256, 256, 0, stream>>>(W_qkv, wqt, DIMX, 3 * DIMX);
  transpose_bf16<<<(DIMX * DIMX + 255) / 256, 256, 0, stream>>>(W_out, wot, DIMX, DIMX);

  // 2) QKV GEMM: 785 M-groups x 36 N-groups, 4 wave-groups per block
  qkv_gemm<<<(785 * 36) / 4, 128, 0, stream>>>(xb, wqt, b_qkv, qf, qb, kT, vf, vT);

  // 3) zero vT pad columns
  pad_vt<<<(BATCH * HEADS * HDIM * (NPAD - NTOK) + 255) / 256, 256, 0, stream>>>(vT);

  // 4) softmax over tokens (24576 rows, 8 rows/block)
  softmax_k<<<(BATCH * HEADS * HDIM) / 8, 256, 0, stream>>>(kT, ksm);

  // 5) per-(b,h) factor GEMM -> attn^T
  attn_gemm<<<BATCH * HEADS, 256, 0, stream>>>(ksm, vT, attnT);

  // 6) CRPE depthwise convolutions
  {
    int n = BATCH * HEADS * HDIM * GH * GW;
    crpe_conv<<<(n + 255) / 256, 256, 0, stream>>>(vf, cw3, cb3, cw5, cb5, cw7, cb7, rp);
  }

  // 7) out1 = q_scaled @ attn + q*rp  (300 tiles per (b,h), 8 wave-tiles/block)
  out1_gemm<<<dim3((300 + 7) / 8, BATCH * HEADS), 256, 0, stream>>>(qb, attnT, qf, rp, o1);

  // 8) final projection: 785 M-groups x 12 N-groups
  proj_gemm<<<(785 * 12) / 4, 128, 0, stream>>>(o1, wot, b_out, y);
}
